// SparseSelfAttention_30906584662593
// MI455X (gfx1250) — compile-verified, matrix operand order checked by reading, not run on hardware
//
#include <hip/hip_runtime.h>

typedef __attribute__((ext_vector_type(16))) _Float16 v16h;
typedef __attribute__((ext_vector_type(8)))  _Float16 v8h;
typedef __attribute__((ext_vector_type(4)))  _Float16 v4h;
typedef __attribute__((ext_vector_type(8)))  float    v8f;
typedef __attribute__((ext_vector_type(4)))  float    v4f;

#define DM   1024      // d_model
#define SEQ  4096
#define BATCH 2
#define NROWS (BATCH*SEQ)   // 8192

#define WMMA_F16(a,b,c) __builtin_amdgcn_wmma_f32_16x16x32_f16(false,(a),false,(b),(short)0,(c),false,false)

// Load a 16x32 fragment (A-layout, 16-bit) from an f32 row-major matrix.
// lane<16  : row = base+r, K in {0..7, 16..23}
// lane>=16 : row = base+r, K in {8..15, 24..31}
__device__ __forceinline__ v16h load_frag_f32(const float* __restrict__ base,
                                              int row, int k0, int hl) {
    const float* p = base + (size_t)row * DM + k0 + hl * 8;
    v4f a0 = *(const v4f*)(p);
    v4f a1 = *(const v4f*)(p + 4);
    v4f a2 = *(const v4f*)(p + 16);
    v4f a3 = *(const v4f*)(p + 20);
    v4h h0 = __builtin_convertvector(a0, v4h);
    v4h h1 = __builtin_convertvector(a1, v4h);
    v4h h2 = __builtin_convertvector(a2, v4h);
    v4h h3 = __builtin_convertvector(a3, v4h);
    v8h lo = __builtin_shufflevector(h0, h1, 0,1,2,3,4,5,6,7);
    v8h hi = __builtin_shufflevector(h2, h3, 0,1,2,3,4,5,6,7);
    return __builtin_shufflevector(lo, hi, 0,1,2,3,4,5,6,7,8,9,10,11,12,13,14,15);
}

// Same fragment pattern but from an f16 row-major matrix.
__device__ __forceinline__ v16h load_frag_f16(const _Float16* __restrict__ base,
                                              size_t row, int col0, int k0, int hl) {
    const _Float16* p = base + row * DM + col0 + k0 + hl * 8;
    v8h lo = *(const v8h*)(p);
    v8h hi = *(const v8h*)(p + 16);
    return __builtin_shufflevector(lo, hi, 0,1,2,3,4,5,6,7,8,9,10,11,12,13,14,15);
}

// ---------------------------------------------------------------------------
// Kernel 1: fused Q/K/V projection GEMM.  y = x @ W^T + b  (f32 in, f16 out)
// Block = 256 threads = 8 waves; block tile 256(M) x 128(N); wave tile 64x64.
// grid = (NROWS/256, DM/128, 3)   (z selects which projection)
// ---------------------------------------------------------------------------
__global__ __launch_bounds__(256) void proj_gemm_kernel(
    const float* __restrict__ q_in, const float* __restrict__ k_in, const float* __restrict__ v_in,
    const float* __restrict__ Wq, const float* __restrict__ bq,
    const float* __restrict__ Wk, const float* __restrict__ bk,
    const float* __restrict__ Wv, const float* __restrict__ bv,
    _Float16* __restrict__ Qh, _Float16* __restrict__ Kh, _Float16* __restrict__ Vh)
{
    const int z = blockIdx.z;
    const float* __restrict__ X    = (z == 0) ? q_in : (z == 1) ? k_in : v_in;
    const float* __restrict__ W    = (z == 0) ? Wq   : (z == 1) ? Wk   : Wv;
    const float* __restrict__ bias = (z == 0) ? bq   : (z == 1) ? bk   : bv;
    _Float16* __restrict__ Y       = (z == 0) ? Qh   : (z == 1) ? Kh   : Vh;

    const int wid  = threadIdx.x >> 5;
    const int lane = threadIdx.x & 31;
    const int hl   = lane >> 4;       // half-wave id
    const int r    = lane & 15;

    const int m0 = blockIdx.x * 256 + (wid >> 1) * 64;  // wave's 64-row strip
    const int n0 = blockIdx.y * 128 + (wid & 1) * 64;   // wave's 64-col strip

    v8f acc[4][4];
#pragma unroll
    for (int i = 0; i < 4; ++i)
#pragma unroll
        for (int j = 0; j < 4; ++j) acc[i][j] = (v8f)(0.0f);

#pragma unroll 1
    for (int k0 = 0; k0 < DM; k0 += 32) {
        v16h afrag[4], bfrag[4];
#pragma unroll
        for (int im = 0; im < 4; ++im)
            afrag[im] = load_frag_f32(X, m0 + im * 16 + r, k0, hl);
#pragma unroll
        for (int in = 0; in < 4; ++in)
            bfrag[in] = load_frag_f32(W, n0 + in * 16 + r, k0, hl);   // B col n == W row n
#pragma unroll
        for (int im = 0; im < 4; ++im)
#pragma unroll
            for (int in = 0; in < 4; ++in)
                acc[im][in] = WMMA_F16(afrag[im], bfrag[in], acc[im][in]);
    }

    // C layout: VGPR v, lane<16 -> (M=v, N=r); lane>=16 -> (M=8+v, N=r)
#pragma unroll
    for (int im = 0; im < 4; ++im) {
#pragma unroll
        for (int in = 0; in < 4; ++in) {
            const int col = n0 + in * 16 + r;
            const float bv_ = bias[col];
#pragma unroll
            for (int v = 0; v < 8; ++v) {
                const int row = m0 + im * 16 + hl * 8 + v;
                Y[(size_t)row * DM + col] = (_Float16)(acc[im][in][v] + bv_);
            }
        }
    }
}

// ---------------------------------------------------------------------------
// Kernel 2: block-diagonal attention.  One wave handles TWO consecutive 8-row
// blocks (a 16x64 tile) of one head: scores via 2 WMMAs (K=64), per-row
// softmax via 8-lane shuffles, P routed through LDS, P@V via 4 WMMAs.
// grid = (B*H*(nb/2))/8 blocks of 256 threads.
// ---------------------------------------------------------------------------
__global__ __launch_bounds__(256) void block_attn_kernel(
    const _Float16* __restrict__ Qh, const _Float16* __restrict__ Kh,
    const _Float16* __restrict__ Vh, float* __restrict__ out)
{
    __shared__ __align__(16) _Float16 pbuf[8][16][16];   // per-wave 16x16 P tile

    const int wid  = threadIdx.x >> 5;
    const int lane = threadIdx.x & 31;
    const int hl   = lane >> 4;
    const int r    = lane & 15;

    const int pairIdx = blockIdx.x * 8 + wid;   // 0 .. B*16*256-1
    const int b  = pairIdx >> 12;               // 16 heads * 256 pairs = 4096 per batch
    const int h  = (pairIdx >> 8) & 15;
    const int p  = pairIdx & 255;
    const size_t rowBase = (size_t)b * SEQ + (size_t)p * 16;
    const int col0 = h * 64;

    // ---- scores S = Q16 (16x64) x K16^T : two chained WMMAs over K=64 ----
    v8f s = (v8f)(0.0f);
    {
        v16h qa0 = load_frag_f16(Qh, rowBase + r, col0, 0,  hl);
        v16h kb0 = load_frag_f16(Kh, rowBase + r, col0, 0,  hl);  // B col n == K row n
        v16h qa1 = load_frag_f16(Qh, rowBase + r, col0, 32, hl);
        v16h kb1 = load_frag_f16(Kh, rowBase + r, col0, 32, hl);
        s = WMMA_F16(qa0, kb0, s);
        s = WMMA_F16(qa1, kb1, s);
    }

    // ---- masked softmax per row (8 valid columns per row) ----
    // lane holds (M = hl*8+v, N = r); intra-block iff (M<8)==(N<8):
    const bool valid = (hl == 0) ? (r < 8) : (r >= 8);
#pragma unroll
    for (int v = 0; v < 8; ++v) {
        float x = s[v] * 0.125f;                      // 1/sqrt(dk), dk=64
        float m = x;
#pragma unroll
        for (int off = 1; off < 8; off <<= 1)
            m = fmaxf(m, __shfl_xor(m, off, 8));      // reduce over the 8-lane group
        float e = __expf(x - m);
        float t = e;
#pragma unroll
        for (int off = 1; off < 8; off <<= 1)
            t += __shfl_xor(t, off, 8);
        float pn = e / t;
        // full 16x16 tile is covered exactly once across lanes; off-diagonal -> 0
        pbuf[wid][hl * 8 + v][r] = (_Float16)(valid ? pn : 0.0f);
    }

    asm volatile("s_wait_dscnt 0" ::: "memory");      // wave-local LDS RAW fence

    // ---- reload P as a 16x32 A-fragment (K 16->32 zero-padded) ----
    v16h pf = (v16h)((_Float16)0.0f);
    {
        v8h pl = *(const v8h*)&pbuf[wid][r][hl * 8];
#pragma unroll
        for (int j = 0; j < 8; ++j) pf[j] = pl[j];
    }

    // ---- O = P (16x16) x V16 (16x64) : 4 WMMAs over N-tiles ----
#pragma unroll
    for (int t = 0; t < 4; ++t) {
        const int col = col0 + t * 16 + r;
        v16h vf = (v16h)((_Float16)0.0f);             // B frag: K rows 16..31 are zero-pad
        const _Float16* vp = Vh + (rowBase + hl * 8) * DM + col;
#pragma unroll
        for (int j = 0; j < 8; ++j) vf[j] = vp[(size_t)j * DM];

        v8f o = (v8f)(0.0f);
        o = WMMA_F16(pf, vf, o);
#pragma unroll
        for (int v = 0; v < 8; ++v) {
            const size_t row = rowBase + hl * 8 + v;
            out[row * DM + col] = o[v];
        }
    }
}

// ---------------------------------------------------------------------------
extern "C" void kernel_launch(void* const* d_in, const int* in_sizes, int n_in,
                              void* d_out, int out_size, void* d_ws, size_t ws_size,
                              hipStream_t stream) {
    const float* query = (const float*)d_in[0];
    const float* key   = (const float*)d_in[1];
    const float* value = (const float*)d_in[2];
    const float* Wq    = (const float*)d_in[3];
    const float* bq    = (const float*)d_in[4];
    const float* Wk    = (const float*)d_in[5];
    const float* bk    = (const float*)d_in[6];
    const float* Wv    = (const float*)d_in[7];
    const float* bv    = (const float*)d_in[8];

    const size_t projBytes = (size_t)NROWS * DM * sizeof(_Float16);   // 16 MB each
    _Float16* Qh = (_Float16*)d_ws;
    _Float16* Kh = (_Float16*)((char*)d_ws + projBytes);
    _Float16* Vh = (_Float16*)((char*)d_ws + 2 * projBytes);

    dim3 gGemm(NROWS / 256, DM / 128, 3);
    proj_gemm_kernel<<<gGemm, 256, 0, stream>>>(query, key, value,
                                                Wq, bq, Wk, bk, Wv, bv,
                                                Qh, Kh, Vh);

    const int nPairs = BATCH * 16 * (SEQ / 16);   // 8192 wave-tasks
    block_attn_kernel<<<nPairs / 8, 256, 0, stream>>>(Qh, Kh, Vh, (float*)d_out);
}
